// GSoP_77558519431429
// MI455X (gfx1250) — compile-verified, matrix operand order checked by reading, not run on hardware
//
#include <hip/hip_runtime.h>
#include <hip/hip_bf16.h>
#include <stdint.h>

// ---------------- problem constants ----------------
constexpr int B_ = 32, C_ = 512, Ww_ = 56, M_ = 3136, D_ = 128, OD_ = 256;
constexpr float EPSV = 1e-5f;

// ---------------- WMMA vector types ----------------
typedef __attribute__((ext_vector_type(16))) _Float16 hvec16;
typedef __attribute__((ext_vector_type(8)))  _Float16 hvec8;
typedef __attribute__((ext_vector_type(8)))  float    fvec8;

union FragU { hvec16 v; hvec8 h[2]; };

// Gather one 16x32 f16 fragment (A layout per CDNA5 ISA; B uses mirrored layout
// with the tile stored [col][K] in LDS so the same gather applies).
__device__ inline hvec16 frag16(const _Float16* __restrict__ base, int half) {
    FragU f;
    f.h[0] = *(const hvec8*)(base + half * 8);
    f.h[1] = *(const hvec8*)(base + 16 + half * 8);
    return f.v;
}

__device__ inline fvec8 wmma_f16(hvec16 a, hvec16 b, fvec8 c) {
    return __builtin_amdgcn_wmma_f32_16x16x32_f16(false, a, false, b, (short)0, c, false, false);
}

__device__ inline float sigmoidf_(float x) { return 1.0f / (1.0f + __expf(-x)); }

// ---------------- Tensor Data Mover (gfx1250) -------------------------------
#if __has_builtin(__builtin_amdgcn_tensor_load_to_lds)
#define HAVE_TDM 1
#else
#define HAVE_TDM 0
#endif

#if HAVE_TDM
typedef __attribute__((ext_vector_type(4))) unsigned int uvec4;
typedef __attribute__((ext_vector_type(8))) int          ivec8;
typedef __attribute__((ext_vector_type(4))) int          ivec4;

// 2D f16 tile DMA: tile (tile0 x tile1 elements) of a (tdim0 x tdim1) row-major
// tensor with row stride stride0 (elements) -> packed rows in LDS at lds_off.
__device__ inline void tdm_load_2d_f16(uint32_t lds_off, const _Float16* gptr,
                                       uint32_t tdim0, uint32_t tdim1,
                                       uint32_t tile0, uint32_t tile1,
                                       uint32_t stride0) {
    uint64_t ga = (uint64_t)(uintptr_t)gptr;
    uvec4 g0;
    g0[0] = 1u;                                                 // count=1 (valid D#)
    g0[1] = lds_off;                                            // lds_addr (bytes)
    g0[2] = (uint32_t)ga;                                       // global_addr[31:0]
    g0[3] = (uint32_t)((ga >> 32) & 0x01FFFFFFu) | 0x80000000u; // [56:32] | type=2
    ivec8 g1;
    g1[0] = (int)(1u << 16);                          // data_size=1 -> 2 bytes
    g1[1] = (int)(tdim0 << 16);                       // tensor_dim0[15:0]
    g1[2] = (int)((tdim0 >> 16) | (tdim1 << 16));     // dim0 hi | dim1 lo
    g1[3] = (int)((tdim1 >> 16) | (tile0 << 16));     // dim1 hi | tile_dim0
    g1[4] = (int)(tile1 & 0xFFFFu);                   // tile_dim1 (tile_dim2=0)
    g1[5] = (int)stride0;                             // tensor_dim0_stride[31:0]
    g1[6] = 0;
    g1[7] = 0;
    ivec4 zg = {0, 0, 0, 0};
#if defined(__clang_major__) && (__clang_major__ >= 23)
    ivec8 zg3 = {0, 0, 0, 0, 0, 0, 0, 0};
    __builtin_amdgcn_tensor_load_to_lds(g0, g1, zg, zg, zg3, 0);
#else
    __builtin_amdgcn_tensor_load_to_lds(g0, g1, zg, zg, 0);
#endif
}
#endif // HAVE_TDM

// ---------------- kernel 0: pack weights to f16 (256 x 512) ----------------
__global__ __launch_bounds__(256) void k_wconv(const float* __restrict__ wdr,
                                               const float* __restrict__ wdrs,
                                               _Float16* __restrict__ Wh) {
    int i = blockIdx.x * 256 + threadIdx.x; // 0 .. 131071
    int o = i >> 9, c = i & 511;
    float v = (o < D_) ? wdr[o * C_ + c] : wdrs[(o - D_) * C_ + c];
    Wh[i] = (_Float16)v;
}

// ---------------- kernel 1: fused dual 1x1 conv (256x512x3136 GEMM/batch) ----
// grid (49, 32): 64 output pixels per block, all 256 output channels.
__global__ __launch_bounds__(256) void k_conv(
    const float* __restrict__ x, const _Float16* __restrict__ Wh,
    const float* __restrict__ bdr, const float* __restrict__ g1,
    const float* __restrict__ be1, const float* __restrict__ m1,
    const float* __restrict__ v1,
    const float* __restrict__ bdrs, const float* __restrict__ g3,
    const float* __restrict__ be3, const float* __restrict__ m3,
    const float* __restrict__ v3,
    _Float16* __restrict__ yh, _Float16* __restrict__ sh) {
    __shared__ __align__(16) _Float16 lA[OD_ * 32]; // weights   [o][k]     16 KB
    __shared__ __align__(16) _Float16 lB[64 * 32];  // relu(x)^T [m][k]      4 KB

    const int t = threadIdx.x;
    const int wv = t >> 5, lane = t & 31, lrow = lane & 15, lhalf = lane >> 4;
    const int bi = blockIdx.y;
    const int m0 = blockIdx.x * 64;
    const float* xb = x + (size_t)bi * C_ * M_;

    fvec8 acc[2][4] = {};

    for (int kc = 0; kc < C_ / 32; ++kc) {
        const int k0 = kc * 32;
        // stage A (weights 256x32) via Tensor Data Mover when available
#if HAVE_TDM
        if (wv == 0)
            tdm_load_2d_f16((uint32_t)(uintptr_t)lA, Wh + k0,
                            C_, OD_, 32, OD_, C_);
#else
        {
            const uint4* src = (const uint4*)(Wh + (size_t)t * C_ + k0);
            uint4* dst = (uint4*)(lA + t * 32);
#pragma unroll
            for (int i = 0; i < 4; ++i) dst[i] = src[i];
        }
#endif
        // stage B: 32 channels x 64 pixels via float4, relu + cvt, store [m][k]
#pragma unroll
        for (int i = 0; i < 2; ++i) {
            int lin4 = t + 256 * i;          // 0..511 float4s
            int cl = lin4 >> 4, mq = lin4 & 15;
            float4 v = *(const float4*)(xb + (size_t)(k0 + cl) * M_ + m0 + mq * 4);
            lB[(mq * 4 + 0) * 32 + cl] = (_Float16)fmaxf(v.x, 0.0f);
            lB[(mq * 4 + 1) * 32 + cl] = (_Float16)fmaxf(v.y, 0.0f);
            lB[(mq * 4 + 2) * 32 + cl] = (_Float16)fmaxf(v.z, 0.0f);
            lB[(mq * 4 + 3) * 32 + cl] = (_Float16)fmaxf(v.w, 0.0f);
        }
        if (kc + 1 < C_ / 32)
            __builtin_prefetch(xb + (size_t)(k0 + 32) * M_ + m0 + (t & 63), 0, 1);
#if HAVE_TDM
        if (wv == 0) __builtin_amdgcn_s_wait_tensorcnt(0);
#endif
        __syncthreads();

        hvec16 af[2], bf[4];
#pragma unroll
        for (int rt = 0; rt < 2; ++rt)
            af[rt] = frag16(lA + (wv * 32 + rt * 16 + lrow) * 32, lhalf);
#pragma unroll
        for (int ct = 0; ct < 4; ++ct)
            bf[ct] = frag16(lB + (ct * 16 + lrow) * 32, lhalf);
#pragma unroll
        for (int rt = 0; rt < 2; ++rt)
#pragma unroll
            for (int ct = 0; ct < 4; ++ct)
                acc[rt][ct] = wmma_f16(af[rt], bf[ct], acc[rt][ct]);
        __syncthreads();
    }

    // epilogue: bias + BN (+relu for y), write f16
#pragma unroll
    for (int rt = 0; rt < 2; ++rt)
#pragma unroll
        for (int ct = 0; ct < 4; ++ct)
#pragma unroll
            for (int e = 0; e < 8; ++e) {
                int o = wv * 32 + rt * 16 + e + lhalf * 8;
                int m = m0 + ct * 16 + lrow;
                float v = acc[rt][ct][e];
                if (o < D_) {
                    v += bdr[o];
                    float sc = g1[o] * rsqrtf(v1[o] + EPSV);
                    v = fmaxf((v - m1[o]) * sc + be1[o], 0.0f);
                    yh[((size_t)bi * D_ + o) * M_ + m] = (_Float16)v;
                } else {
                    int od = o - D_;
                    v += bdrs[od];
                    float sc = g3[od] * rsqrtf(v3[od] + EPSV);
                    v = (v - m3[od]) * sc + be3[od];
                    sh[((size_t)bi * D_ + od) * M_ + m] = (_Float16)v;
                }
            }
}

// ---------------- kernel 2: per-(b,d) mean of y and 8x8 pooling of s --------
__global__ __launch_bounds__(256) void k_stats(const _Float16* __restrict__ yh,
                                               const _Float16* __restrict__ sh,
                                               float* __restrict__ muy,
                                               float* __restrict__ sp) {
    __shared__ float cell[64];
    __shared__ float red[256];
    const int t = threadIdx.x;
    const int bd = blockIdx.x;
    if (t < 64) cell[t] = 0.0f;
    __syncthreads();

    float sum = 0.0f;
    for (int m = t; m < M_; m += 256) {
        sum += (float)yh[(size_t)bd * M_ + m];
        float sv = (float)sh[(size_t)bd * M_ + m];
        int h = m / Ww_, w = m % Ww_;
        atomicAdd(&cell[(h / 7) * 8 + (w / 7)], sv);  // ds_add_f32
    }
    red[t] = sum;
    __syncthreads();
    for (int s = 128; s > 0; s >>= 1) {
        if (t < s) red[t] += red[t + s];
        __syncthreads();
    }
    if (t == 0) muy[bd] = red[0] * (1.0f / (float)M_);
    if (t < 64) sp[(size_t)bd * 64 + t] = cell[t] * (1.0f / 49.0f);
}

// ---------------- kernel 3: per-batch 128x128 covariance (y yT) via WMMA ----
// TDM double-buffered: chunk k+1 DMAs into the other 8 KB buffer while all
// 8 waves stream WMMAs out of chunk k.
__global__ __launch_bounds__(256) void k_covy(const _Float16* __restrict__ yh,
                                              const float* __restrict__ muy,
                                              const float* __restrict__ g2,
                                              const float* __restrict__ be2,
                                              const float* __restrict__ m2,
                                              const float* __restrict__ v2,
                                              float* __restrict__ covf) {
    __shared__ __align__(16) _Float16 lY[2][D_ * 32]; // 2 x 8 KB
    const int t = threadIdx.x;
    const int wv = t >> 5, lane = t & 31, lrow = lane & 15, lhalf = lane >> 4;
    const int bi = blockIdx.x;
    const _Float16* yb = yh + (size_t)bi * D_ * M_;
    constexpr int NK = M_ / 32; // 98

    fvec8 acc[8] = {};

#if HAVE_TDM
    const uint32_t lds0 = (uint32_t)(uintptr_t)&lY[0][0];
    const uint32_t lds1 = (uint32_t)(uintptr_t)&lY[1][0];
    if (wv == 0) tdm_load_2d_f16(lds0, yb, M_, D_, 32, D_, M_);
#endif

    for (int kc = 0; kc < NK; ++kc) {
#if HAVE_TDM
        if (wv == 0) {
            if (kc + 1 < NK) {
                tdm_load_2d_f16(((kc + 1) & 1) ? lds1 : lds0,
                                yb + (size_t)(kc + 1) * 32, M_, D_, 32, D_, M_);
                __builtin_amdgcn_s_wait_tensorcnt(1);
            } else {
                __builtin_amdgcn_s_wait_tensorcnt(0);
            }
        }
#else
        {   // 128 rows x 32 halfs; each thread moves 16 halfs (2x uint4)
            int d = t >> 1, part = t & 1;
            const uint4* src = (const uint4*)(yb + (size_t)d * M_ + kc * 32 + part * 16);
            uint4* dst = (uint4*)(&lY[kc & 1][0] + d * 32 + part * 16);
            dst[0] = src[0];
            dst[1] = src[1];
        }
#endif
        __syncthreads();
        const _Float16* buf = &lY[kc & 1][0];
        hvec16 af = frag16(buf + (wv * 16 + lrow) * 32, lhalf);
#pragma unroll
        for (int ct = 0; ct < 8; ++ct) {
            hvec16 bf = frag16(buf + (ct * 16 + lrow) * 32, lhalf);
            acc[ct] = wmma_f16(af, bf, acc[ct]);
        }
        __syncthreads();
    }

    const float* mu = muy + bi * D_;
#pragma unroll
    for (int ct = 0; ct < 8; ++ct)
#pragma unroll
        for (int e = 0; e < 8; ++e) {
            int i = wv * 16 + e + lhalf * 8;
            int j = ct * 16 + lrow;
            float cov = acc[ct][e] * (1.0f / (float)M_) - mu[i] * mu[j];
            float sc = g2[i] * rsqrtf(v2[i] + EPSV);
            covf[((size_t)bi * D_ + i) * D_ + j] = (cov - m2[i]) * sc + be2[i];
        }
}

// ---------------- kernel 4: per-batch 64x64 pooled covariance via WMMA ------
__global__ __launch_bounds__(256) void k_covs(const float* __restrict__ sp,
                                              const float* __restrict__ g4,
                                              const float* __restrict__ be4,
                                              const float* __restrict__ m4,
                                              const float* __restrict__ v4,
                                              float* __restrict__ covsf) {
    __shared__ __align__(16) _Float16 lS[64 * 128]; // [m][c] 16 KB
    __shared__ float csum[64];
    const int t = threadIdx.x;
    const int wv = t >> 5, lane = t & 31, lrow = lane & 15, lhalf = lane >> 4;
    const int bi = blockIdx.x;
    const float* spb = sp + (size_t)bi * D_ * 64;

#pragma unroll
    for (int i = 0; i < 32; ++i) {
        int lin = t + 256 * i;      // 0..8191
        int c = lin >> 6, mm = lin & 63;
        lS[mm * 128 + c] = (_Float16)spb[(size_t)c * 64 + mm];
    }
    __syncthreads();
    if (t < 64) {
        float s = 0.0f;
        for (int c = 0; c < D_; ++c) s += (float)lS[t * 128 + c];
        csum[t] = s;
    }
    __syncthreads();

    const int rt = wv >> 1;             // row tile 0..3
    const int cb = (wv & 1) * 2;        // col tile base
    fvec8 acc[2] = {};
#pragma unroll
    for (int kc = 0; kc < 4; ++kc) {
        int k0 = kc * 32;
        hvec16 af = frag16(lS + (rt * 16 + lrow) * 128 + k0, lhalf);
#pragma unroll
        for (int q = 0; q < 2; ++q) {
            hvec16 bf = frag16(lS + ((cb + q) * 16 + lrow) * 128 + k0, lhalf);
            acc[q] = wmma_f16(af, bf, acc[q]);
        }
    }

#pragma unroll
    for (int q = 0; q < 2; ++q)
#pragma unroll
        for (int e = 0; e < 8; ++e) {
            int mm = rt * 16 + e + lhalf * 8;
            int n = (cb + q) * 16 + lrow;
            float cv = (acc[q][e] - csum[mm] * csum[n] * (1.0f / 64.0f)) * (1.0f / 64.0f);
            float sc = g4[mm] * rsqrtf(v4[mm] + EPSV);
            covsf[((size_t)bi * 64 + mm) * 64 + n] = (cv - m4[mm]) * sc + be4[mm];
        }
}

// ---------------- kernel 5: small attention chains (per batch) --------------
__global__ __launch_bounds__(256) void k_attn(
    const float* __restrict__ covf, const float* __restrict__ wrow,
    const float* __restrict__ brow, const float* __restrict__ wfc,
    const float* __restrict__ bfc,
    const float* __restrict__ covsf, const float* __restrict__ wrows,
    const float* __restrict__ brows, const float* __restrict__ wfcs,
    const float* __restrict__ bfcs,
    float* __restrict__ cattg, float* __restrict__ pattg) {
    __shared__ float vch[512];
    __shared__ float vsb[256];
    const int t = threadIdx.x;
    const int bi = blockIdx.x;

    // vch[i,j] = sum_k cov[b,i,k] * wrow[i,j,k]
    for (int idx = t; idx < 512; idx += 256) {
        int i = idx >> 2;
        const float* cv = covf + ((size_t)bi * D_ + i) * D_;
        const float* wr = wrow + (size_t)idx * D_;
        float s = brow[idx];
        for (int k = 0; k < D_; ++k) s += cv[k] * wr[k];
        vch[idx] = s;
    }
    // vs[i,j] = relu(sum_k covs[b,i,k] * wrows[i,j,k])
    {
        int i = t >> 2;
        const float* cs = covsf + ((size_t)bi * 64 + i) * 64;
        const float* wr = wrows + (size_t)t * 64;
        float s = brows[t];
        for (int k = 0; k < 64; ++k) s += cs[k] * wr[k];
        vsb[t] = fmaxf(s, 0.0f);
    }
    __syncthreads();
    // catt = sigmoid(vch @ wfc^T + bfc)
    for (int c = t; c < C_; c += 256) {
        const float* wf = wfc + (size_t)c * 512;
        float s = bfc[c];
        for (int k = 0; k < 512; ++k) s += vch[k] * wf[k];
        cattg[(size_t)bi * C_ + c] = sigmoidf_(s);
    }
    // patt = sigmoid(vs @ wfcs^T + bfcs)
    if (t < 64) {
        const float* wf = wfcs + (size_t)t * 256;
        float s = bfcs[t];
        for (int k = 0; k < 256; ++k) s += vsb[k] * wf[k];
        pattg[(size_t)bi * 64 + t] = sigmoidf_(s);
    }
}

// ---------------- kernel 6: out = x*catt + relu(x*patt) ---------------------
__global__ __launch_bounds__(256) void k_final(const float* __restrict__ x,
                                               const float* __restrict__ cattg,
                                               const float* __restrict__ pattg,
                                               float* __restrict__ out) {
    const int64_t n4 = (int64_t)B_ * C_ * (M_ / 4);
    int64_t i4 = (int64_t)blockIdx.x * blockDim.x + threadIdx.x;
    const int64_t stride = (int64_t)gridDim.x * blockDim.x;
    for (; i4 < n4; i4 += stride) {
        int64_t lin = i4 * 4;
        int m = (int)(lin % M_);
        int64_t bc = lin / M_;
        int c = (int)(bc & (C_ - 1));
        int b = (int)(bc >> 9);
        float ca = cattg[(size_t)b * C_ + c];
        const float* pt = pattg + (size_t)b * 64;
        int h = m / Ww_, w = m % Ww_;     // 56 % 4 == 0 -> h constant over 4
        int pr = (h / 7) * 8;
        float4 xv = *(const float4*)(x + lin);
        float4 ov;
        {
            float p = pt[pr + (w + 0) / 7];
            ov.x = xv.x * ca + fmaxf(xv.x * p, 0.0f);
        }
        {
            float p = pt[pr + (w + 1) / 7];
            ov.y = xv.y * ca + fmaxf(xv.y * p, 0.0f);
        }
        {
            float p = pt[pr + (w + 2) / 7];
            ov.z = xv.z * ca + fmaxf(xv.z * p, 0.0f);
        }
        {
            float p = pt[pr + (w + 3) / 7];
            ov.w = xv.w * ca + fmaxf(xv.w * p, 0.0f);
        }
        *(float4*)(out + lin) = ov;
    }
}

// ---------------- host launcher ---------------------------------------------
extern "C" void kernel_launch(void* const* d_in, const int* in_sizes, int n_in,
                              void* d_out, int out_size, void* d_ws, size_t ws_size,
                              hipStream_t stream) {
    const float* x    = (const float*)d_in[0];
    const float* wdr  = (const float*)d_in[1];
    const float* bdr  = (const float*)d_in[2];
    const float* g1   = (const float*)d_in[3];
    const float* be1  = (const float*)d_in[4];
    const float* m1   = (const float*)d_in[5];
    const float* v1   = (const float*)d_in[6];
    const float* g2   = (const float*)d_in[7];
    const float* be2  = (const float*)d_in[8];
    const float* m2   = (const float*)d_in[9];
    const float* v2   = (const float*)d_in[10];
    const float* wrow = (const float*)d_in[11];
    const float* brow = (const float*)d_in[12];
    const float* wfc  = (const float*)d_in[13];
    const float* bfc  = (const float*)d_in[14];
    const float* wdrs = (const float*)d_in[15];
    const float* bdrs = (const float*)d_in[16];
    const float* g3   = (const float*)d_in[17];
    const float* be3  = (const float*)d_in[18];
    const float* m3   = (const float*)d_in[19];
    const float* v3   = (const float*)d_in[20];
    const float* g4   = (const float*)d_in[21];
    const float* be4  = (const float*)d_in[22];
    const float* m4   = (const float*)d_in[23];
    const float* v4   = (const float*)d_in[24];
    const float* wrows= (const float*)d_in[25];
    const float* brows= (const float*)d_in[26];
    const float* wfcs = (const float*)d_in[27];
    const float* bfcs = (const float*)d_in[28];
    float* out = (float*)d_out;

    char* ws = (char*)d_ws;
    size_t off = 0;
    auto carve = [&](size_t bytes) {
        char* p = ws + off;
        off += (bytes + 255) & ~(size_t)255;
        return p;
    };
    _Float16* Wh   = (_Float16*)carve((size_t)OD_ * C_ * 2);       // 256 KB
    _Float16* yh   = (_Float16*)carve((size_t)B_ * D_ * M_ * 2);   // ~24.5 MB
    _Float16* sh   = (_Float16*)carve((size_t)B_ * D_ * M_ * 2);   // ~24.5 MB
    float*    muy  = (float*)carve((size_t)B_ * D_ * 4);
    float*    sp   = (float*)carve((size_t)B_ * D_ * 64 * 4);      // 1 MB
    float*    covf = (float*)carve((size_t)B_ * D_ * D_ * 4);      // 2 MB
    float*    covsf= (float*)carve((size_t)B_ * 64 * 64 * 4);      // 0.5 MB
    float*    cattg= (float*)carve((size_t)B_ * C_ * 4);
    float*    pattg= (float*)carve((size_t)B_ * 64 * 4);
    (void)ws_size; (void)in_sizes; (void)n_in; (void)out_size;

    k_wconv<<<(OD_ * C_) / 256, 256, 0, stream>>>(wdr, wdrs, Wh);
    k_conv<<<dim3(M_ / 64, B_), 256, 0, stream>>>(x, Wh,
        bdr, g1, be1, m1, v1, bdrs, g3, be3, m3, v3, yh, sh);
    k_stats<<<B_ * D_, 256, 0, stream>>>(yh, sh, muy, sp);
    k_covy<<<B_, 256, 0, stream>>>(yh, muy, g2, be2, m2, v2, covf);
    k_covs<<<B_, 256, 0, stream>>>(sp, g4, be4, m4, v4, covsf);
    k_attn<<<B_, 256, 0, stream>>>(covf, wrow, brow, wfc, bfc,
                                   covsf, wrows, brows, wfcs, bfcs, cattg, pattg);
    k_final<<<8192, 256, 0, stream>>>(x, cattg, pattg, out);
}
